// SAKT2_24970939859704
// MI455X (gfx1250) — matrix operand order repaired and checked by reading, not compiled, not run on hardware
//
#include <hip/hip_runtime.h>
#include <hip/hip_bf16.h>
#include <stdint.h>

typedef _Float16 half_t;
typedef __attribute__((ext_vector_type(16))) _Float16 v16h;
typedef __attribute__((ext_vector_type(8)))  _Float16 v8h;
typedef __attribute__((ext_vector_type(8)))  float    v8f;

#define DEV __device__ __forceinline__

// ---------------- WMMA fragment helpers (CDNA5 16x16x32 f16 layouts) -------
// A (16x32, f16): lane l holds row (l&15); halves 0-7 = K k0..k0+7,
// halves 8-15 = K k0+16..k0+23, k0 = (l>=16)*8  -> two 16B loads.
DEV v16h load_a_frag(const half_t* __restrict__ base, int lda) {
  int lane = threadIdx.x & 31;
  const half_t* p = base + (size_t)(lane & 15) * lda + ((lane >> 4) << 3);
  v8h lo = *(const v8h*)p;
  v8h hi = *(const v8h*)(p + 16);
  v16h a;
#pragma unroll
  for (int i = 0; i < 8; ++i) { a[i] = lo[i]; a[i + 8] = hi[i]; }
  return a;
}
// B (32x16, f16) where B = Wrow^T: lane l holds col n=(l&15); halves 0-7 =
// K kb..kb+7, halves 8-15 = K kb+16..kb+23, kb = (l>=16)*8 (same as A).
DEV v16h load_b_frag(const half_t* __restrict__ base, int ldb) {
  int lane = threadIdx.x & 31;
  const half_t* p = base + (size_t)(lane & 15) * ldb + ((lane >> 4) << 3);
  v8h lo = *(const v8h*)p;
  v8h hi = *(const v8h*)(p + 16);
  v16h b;
#pragma unroll
  for (int i = 0; i < 8; ++i) { b[i] = lo[i]; b[i + 8] = hi[i]; }
  return b;
}
DEV v8f wmma16(v16h a, v16h b, v8f c) {
  return __builtin_amdgcn_wmma_f32_16x16x32_f16(false, a, false, b, (short)0, c,
                                                false, false);
}
DEV float hred16_max(float v) {            // reduce over the 16-lane half-wave
#pragma unroll
  for (int d = 1; d < 16; d <<= 1) v = fmaxf(v, __shfl_xor(v, d, 32));
  return v;
}
DEV float hred16_sum(float v) {
#pragma unroll
  for (int d = 1; d < 16; d <<= 1) v += __shfl_xor(v, d, 32);
  return v;
}
DEV float wred32_sum(float v) {
#pragma unroll
  for (int d = 1; d < 32; d <<= 1) v += __shfl_xor(v, d, 32);
  return v;
}

// ---------------- small utility kernels ------------------------------------
__global__ void cvt_f16_kernel(const float* __restrict__ s, half_t* __restrict__ d, int n) {
  for (int i = blockIdx.x * blockDim.x + threadIdx.x; i < n; i += gridDim.x * blockDim.x)
    d[i] = (half_t)s[i];
}

// Embedding gather: M_hat = M_emb[assess + target*10000], E_hat = E_emb[next]
__global__ void gather_kernel(const int* __restrict__ cate, const int* __restrict__ targ,
                              const float* __restrict__ Memb, const float* __restrict__ Eemb,
                              half_t* __restrict__ M16, half_t* __restrict__ E16) {
  int row = blockIdx.x;              // b*1024 + s
  int b = row >> 10, s = row & 1023;
  int a  = cate[b * 1025 + s];
  int nx = cate[b * 1025 + s + 1];
  int it = targ[b * 1025 + s];
  size_t y = (size_t)a + (size_t)it * 10000u;
  const float* mr = Memb + y * 512;
  const float* er = Eemb + (size_t)nx * 512;
  for (int i = threadIdx.x; i < 512; i += blockDim.x) {
    M16[(size_t)row * 512 + i] = (half_t)mr[i];
    E16[(size_t)row * 512 + i] = (half_t)er[i];
  }
}

// ---------------- generic f16 WMMA GEMM: C = act(A * W^T + bias) -----------
// A: (M,K) f16 row-major. W: (N,K) f16 row-major. fp32 accumulate.
enum { GEMM_RELU = 1, GEMM_OUT_F32 = 2, GEMM_OUT_F16 = 4, GEMM_OUT_F16T = 8 };

__global__ __launch_bounds__(256) void gemm_wmma_kernel(
    const half_t* __restrict__ A, const half_t* __restrict__ W,
    const float* __restrict__ bias, float* __restrict__ Cf,
    half_t* __restrict__ Ch, int M, int N, int K, int ldct, int mode) {
  int wave = threadIdx.x >> 5, lane = threadIdx.x & 31;
  int mb = M >> 6;
  int mblk = blockIdx.x % mb, nblk = blockIdx.x / mb;
  int wm = wave & 3, wn = wave >> 2;                 // 4 m-tiles x 2 col-groups
  int row0 = (mblk << 6) + (wm << 4);
  int col0 = (nblk << 6) + (wn << 5);
  v8f acc0 = {}, acc1 = {};
  if (bias) {
    float b0 = bias[col0 + (lane & 15)];
    float b1 = bias[col0 + 16 + (lane & 15)];
#pragma unroll
    for (int v = 0; v < 8; ++v) { acc0[v] = b0; acc1[v] = b1; }
  }
  const half_t* Ap = A + (size_t)row0 * K;
  const half_t* W0p = W + (size_t)col0 * K;
  const half_t* W1p = W + (size_t)(col0 + 16) * K;
  const half_t* PfA = Ap + (size_t)(lane & 15) * K;
  const half_t* PfW = W0p + (size_t)(lane & 15) * K;
  for (int kk = 0; kk < K; kk += 32) {
    // Unconditional +1-tile speculative prefetch (branch-free mainloop;
    // overshoot past K is harmless: speculative, dropped on fault).
    __builtin_prefetch(PfA + kk + 32, 0, 1);
    __builtin_prefetch(PfW + kk + 32, 0, 1);
    v16h a  = load_a_frag(Ap + kk, K);
    v16h b0 = load_b_frag(W0p + kk, K);
    v16h b1 = load_b_frag(W1p + kk, K);
    acc0 = wmma16(a, b0, acc0);
    acc1 = wmma16(a, b1, acc1);
  }
  if (mode & GEMM_RELU) {
#pragma unroll
    for (int v = 0; v < 8; ++v) { acc0[v] = fmaxf(acc0[v], 0.f); acc1[v] = fmaxf(acc1[v], 0.f); }
  }
  int ccol = lane & 15;
  int crow0 = row0 + ((lane >> 4) << 3);
#pragma unroll
  for (int v = 0; v < 8; ++v) {
    int r = crow0 + v;
    if (mode & GEMM_OUT_F32) {
      Cf[(size_t)r * N + col0 + ccol]      = acc0[v];
      Cf[(size_t)r * N + col0 + 16 + ccol] = acc1[v];
    }
    if (mode & GEMM_OUT_F16) {
      Ch[(size_t)r * N + col0 + ccol]      = (half_t)acc0[v];
      Ch[(size_t)r * N + col0 + 16 + ccol] = (half_t)acc1[v];
    }
    if (mode & GEMM_OUT_F16T) {            // transposed store (for V^T)
      Ch[(size_t)(col0 + ccol) * ldct + r]      = (half_t)acc0[v];
      Ch[(size_t)(col0 + 16 + ccol) * ldct + r] = (half_t)acc1[v];
    }
  }
}

// ---------------- flash attention: one wave per (b, head, 16-query tile) ---
// Q,K: (32768,512) f16 [head-major halves]; Vt: (512,32768) f16 transposed.
__global__ __launch_bounds__(128) void attn_kernel(
    const half_t* __restrict__ Q, const half_t* __restrict__ Kb,
    const half_t* __restrict__ Vt, half_t* __restrict__ AO) {
  __shared__ half_t plds[4 * 16 * 32];
  int wave = threadIdx.x >> 5, lane = threadIdx.x & 31;
  int gidx = blockIdx.x * 4 + wave;     // 0..4095
  int qt = gidx & 63, h = (gidx >> 6) & 1, b = gidx >> 7;
  int qrow0 = b * 1024 + qt * 16;
  const half_t* Qb = Q + (size_t)qrow0 * 512 + h * 256;
  v16h qf[8];
#pragma unroll
  for (int t = 0; t < 8; ++t) qf[t] = load_a_frag(Qb + t * 32, 512);
  v8f o[16];
#pragma unroll
  for (int t = 0; t < 16; ++t) { v8f z = {}; o[t] = z; }
  float m[8], l[8];
#pragma unroll
  for (int v = 0; v < 8; ++v) { m[v] = -1e30f; l[v] = 0.f; }
  half_t* pl = plds + wave * 512;
  int ckey = lane & 15;

  for (int it = 0; it < 32; ++it) {
    int key0 = it * 32;
    v8f s0 = {}, s1 = {};
    const half_t* Kr = Kb + (size_t)(b * 1024 + key0) * 512 + h * 256;
#pragma unroll
    for (int t = 0; t < 8; ++t) {
      v16h bk0 = load_b_frag(Kr + t * 32, 512);
      v16h bk1 = load_b_frag(Kr + (size_t)16 * 512 + t * 32, 512);
      s0 = wmma16(qf[t], bk0, s0);
      s1 = wmma16(qf[t], bk1, s1);
    }
    // reference mask: key k masked unless k >= b-1 (same for every query)
    float mk0 = (key0 + ckey      >= b - 1) ? 0.f : -100000.f;
    float mk1 = (key0 + 16 + ckey >= b - 1) ? 0.f : -100000.f;
    float corr[8];
#pragma unroll
    for (int v = 0; v < 8; ++v) {
      float x0 = s0[v] * 0.0625f + mk0;
      float x1 = s1[v] * 0.0625f + mk1;
      float nm = fmaxf(m[v], hred16_max(fmaxf(x0, x1)));
      float p0 = __expf(x0 - nm), p1 = __expf(x1 - nm);
      float rs = hred16_sum(p0 + p1);
      float c  = __expf(m[v] - nm);
      l[v] = l[v] * c + rs; m[v] = nm; corr[v] = c;
      int prow = v + ((lane >> 4) << 3);
      pl[prow * 32 + ckey]      = (half_t)p0;
      pl[prow * 32 + 16 + ckey] = (half_t)p1;
    }
    asm volatile("s_wait_dscnt 0" ::: "memory");     // LDS transpose RAW
    v16h pf = load_a_frag(pl, 32);
    const half_t* Vb = Vt + (size_t)(h * 256) * 32768 + b * 1024 + key0;
#pragma unroll
    for (int t = 0; t < 16; ++t) {
#pragma unroll
      for (int v = 0; v < 8; ++v) o[t][v] *= corr[v];
      v16h vf = load_b_frag(Vb + (size_t)(t * 16) * 32768, 32768);
      o[t] = wmma16(pf, vf, o[t]);
    }
  }
  int orow0 = qrow0 + ((lane >> 4) << 3);
#pragma unroll
  for (int t = 0; t < 16; ++t)
#pragma unroll
    for (int v = 0; v < 8; ++v)
      AO[(size_t)(orow0 + v) * 512 + h * 256 + t * 16 + ckey] = (half_t)(o[t][v] / l[v]);
}

// ---------------- residual + LayerNorm: one wave per 512-row ---------------
__global__ __launch_bounds__(256) void ln_kernel(
    const float* __restrict__ z, const float* __restrict__ fz,
    const float* __restrict__ g, const float* __restrict__ bb,
    half_t* __restrict__ out) {
  int row = blockIdx.x * 8 + (threadIdx.x >> 5);
  int lane = threadIdx.x & 31;
  const float* zr = z + (size_t)row * 512;
  const float* fr = fz + (size_t)row * 512;
  float x[16], s = 0.f;
#pragma unroll
  for (int i = 0; i < 16; ++i) { x[i] = zr[lane + 32 * i] + fr[lane + 32 * i]; s += x[i]; }
  float mu = wred32_sum(s) * (1.f / 512.f);
  float v = 0.f;
#pragma unroll
  for (int i = 0; i < 16; ++i) { float d = x[i] - mu; v += d * d; }
  float rs = rsqrtf(wred32_sum(v) * (1.f / 512.f) + 1e-5f);
#pragma unroll
  for (int i = 0; i < 16; ++i)
    out[(size_t)row * 512 + lane + 32 * i] =
        (half_t)((x[i] - mu) * rs * g[lane + 32 * i] + bb[lane + 32 * i]);
}

// ---------------- cooperative GRU recurrence -------------------------------
// 16 WGs x 128 thr. Batch(32) is the WMMA M dim; WG g owns h columns
// [g*32, g*32+32). Per step: h staged into LDS via ASYNCcnt-tracked
// global_load_async_to_lds_b128 DMA, gh = h @ Whh^T via WMMA (3 gate accs per
// wave), gate math, h_new to double-buffered global h, device-scope barrier.
__global__ __launch_bounds__(128) void gru_kernel(
    const half_t* __restrict__ gi,    // (32768,1536) f16, rows b*1024+t, + b_ih
    const half_t* __restrict__ Whh,   // (1536,512) f16
    const float* __restrict__ bhh,    // (1536)
    half_t* __restrict__ hbuf,        // 2 * (32*512) f16, zeroed
    half_t* __restrict__ hs,          // (32768,512) f16 out
    unsigned int* __restrict__ cnt) { // 1024 counters, zeroed
  __shared__ half_t hlds[32 * 512];   // 32KB: full h state, shared by 4 waves
  const unsigned int G = 16;
  int wg = blockIdx.x, wave = threadIdx.x >> 5, lane = threadIdx.x & 31;
  int mtile = wave >> 1, ntile = wave & 1;
  int jbase = wg * 32 + ntile * 16;
  int bbase = mtile * 16;
  int j = jbase + (lane & 15);
  float bh_r = bhh[j], bh_z = bhh[512 + j], bh_n = bhh[1024 + j];
  const half_t* Wr = Whh + (size_t)jbase * 512;
  const half_t* Wz = Whh + (size_t)(512 + jbase) * 512;
  const half_t* Wn = Whh + (size_t)(1024 + jbase) * 512;
  unsigned lbase = (unsigned)(uintptr_t)(&hlds[0]);   // LDS aperture offset

  for (int t = 0; t < 1024; ++t) {
    const half_t* hin = hbuf + (size_t)(t & 1) * 16384;
    half_t* hout = hbuf + (size_t)((t + 1) & 1) * 16384;
    // ---- async DMA: stage h_t (32KB) global -> LDS (128 lanes x 16 x b128)
#pragma unroll
    for (int i = 0; i < 16; ++i) {
      unsigned eoff = (unsigned)(threadIdx.x * 8 + i * 1024);   // halves
      unsigned laddr = lbase + eoff * 2;
      unsigned long long gaddr = (unsigned long long)(uintptr_t)(hin + eoff);
      asm volatile("global_load_async_to_lds_b128 %0, %1, off"
                   :: "v"(laddr), "v"(gaddr) : "memory");
    }
    asm volatile("s_wait_asynccnt 0" ::: "memory");
    __syncthreads();
    // ---- gh = h @ Whh^T (3 gate tiles per wave, A-frags from LDS)
    v8f ar = {}, az = {}, an = {};
    for (int kk = 0; kk < 512; kk += 32) {
      v16h a = load_a_frag(hlds + (size_t)bbase * 512 + kk, 512);
      ar = wmma16(a, load_b_frag(Wr + kk, 512), ar);
      az = wmma16(a, load_b_frag(Wz + kk, 512), az);
      an = wmma16(a, load_b_frag(Wn + kk, 512), an);
    }
    int brow0 = bbase + ((lane >> 4) << 3);
#pragma unroll
    for (int v = 0; v < 8; ++v) {
      int bi = brow0 + v;
      size_t gr0 = ((size_t)bi * 1024 + t) * 1536;
      float r = 1.f / (1.f + __expf(-((float)gi[gr0 + j] + ar[v] + bh_r)));
      float zz = 1.f / (1.f + __expf(-((float)gi[gr0 + 512 + j] + az[v] + bh_z)));
      float nn = tanhf((float)gi[gr0 + 1024 + j] + r * (an[v] + bh_n));
      float hp = (float)hlds[(size_t)bi * 512 + j];
      float hn = (1.f - zz) * nn + zz * hp;
      hout[(size_t)bi * 512 + j] = (half_t)hn;
      hs[((size_t)bi * 1024 + t) * 512 + j] = (half_t)hn;
    }
    __threadfence();
    __syncthreads();
    if (threadIdx.x == 0) {
      __hip_atomic_fetch_add(&cnt[t], 1u, __ATOMIC_RELEASE, __HIP_MEMORY_SCOPE_AGENT);
      while (__hip_atomic_load(&cnt[t], __ATOMIC_ACQUIRE, __HIP_MEMORY_SCOPE_AGENT) < G)
        __builtin_amdgcn_s_sleep(1);
    }
    __syncthreads();
  }
}

// ---------------- final head: out[row] = hs_row . Wfin + bfin --------------
__global__ __launch_bounds__(256) void head_kernel(
    const half_t* __restrict__ hs, const float* __restrict__ Wfin,
    const float* __restrict__ bfin, float* __restrict__ out) {
  int row = blockIdx.x * 8 + (threadIdx.x >> 5);
  int lane = threadIdx.x & 31;
  const half_t* hr = hs + (size_t)row * 512;
  float s = 0.f;
  for (int i = lane; i < 512; i += 32) s += (float)hr[i] * Wfin[i];
  s = wred32_sum(s);
  if (lane == 0) out[row] = s + bfin[0];
}

// ---------------------------------------------------------------------------
extern "C" void kernel_launch(void* const* d_in, const int* in_sizes, int n_in,
                              void* d_out, int out_size, void* d_ws, size_t ws_size,
                              hipStream_t stream) {
  (void)in_sizes; (void)n_in; (void)out_size; (void)ws_size;
  const int* cate = (const int*)d_in[0];
  const int* targ = (const int*)d_in[3];
  const float* Memb = (const float*)d_in[4];
  const float* Eemb = (const float*)d_in[5];
  const float* fWq1 = (const float*)d_in[6];
  const float* fWk1 = (const float*)d_in[7];
  const float* fWv1 = (const float*)d_in[8];
  const float* fWq2 = (const float*)d_in[9];
  const float* fWk2 = (const float*)d_in[10];
  const float* fWv2 = (const float*)d_in[11];
  const float* fW0  = (const float*)d_in[12];
  const float* fWf1 = (const float*)d_in[13];
  const float* bf1  = (const float*)d_in[14];
  const float* fWf2 = (const float*)d_in[15];
  const float* bf2  = (const float*)d_in[16];
  const float* ln_g = (const float*)d_in[17];
  const float* ln_b = (const float*)d_in[18];
  const float* fWih0 = (const float*)d_in[19];
  const float* fWhh0 = (const float*)d_in[20];
  const float* bih0  = (const float*)d_in[21];
  const float* bhh0  = (const float*)d_in[22];
  const float* fWih1 = (const float*)d_in[23];
  const float* fWhh1 = (const float*)d_in[24];
  const float* bih1  = (const float*)d_in[25];
  const float* bhh1  = (const float*)d_in[26];
  const float* Wfin  = (const float*)d_in[27];
  const float* bfin  = (const float*)d_in[28];

  char* W = (char*)d_ws;
  size_t off = 0;
  auto take = [&](size_t b) { size_t r = off; off += (b + 255) & ~(size_t)255; return r; };
  const size_t MB = 1024 * 1024;
  size_t oQW = take(512 * 1024), oKW = take(512 * 1024), oVW = take(512 * 1024);
  size_t oW0 = take(512 * 1024), oF1 = take(2 * MB), oF2 = take(2 * MB);
  size_t oIH0 = take(1536 * 512 * 2), oHH0 = take(1536 * 512 * 2);
  size_t oIH1 = take(1536 * 512 * 2), oHH1 = take(1536 * 512 * 2);
  size_t oA = take(160 * MB);       // E16|M16|Q16|K16|Vt  -> later FFH|..|Z16 / GI
  size_t oAO = take(32 * MB);       // attn out -> later z3
  size_t oZ32 = take(64 * MB);      // z fp32   -> later hs0|hs1
  size_t oFZ = take(64 * MB);       // ffn out fp32
  size_t oHB = take(64 * 1024);     // gru h double buffer
  size_t oCNT = take(4096);         // gru step counters

  half_t *QW = (half_t*)(W + oQW), *KW = (half_t*)(W + oKW), *VW = (half_t*)(W + oVW);
  half_t *W0h = (half_t*)(W + oW0), *WF1 = (half_t*)(W + oF1), *WF2 = (half_t*)(W + oF2);
  half_t *WIH0 = (half_t*)(W + oIH0), *WHH0 = (half_t*)(W + oHH0);
  half_t *WIH1 = (half_t*)(W + oIH1), *WHH1 = (half_t*)(W + oHH1);
  half_t *E16 = (half_t*)(W + oA),            *M16 = (half_t*)(W + oA + 32 * MB);
  half_t *Q16 = (half_t*)(W + oA + 64 * MB),  *K16 = (half_t*)(W + oA + 96 * MB);
  half_t *Vt16 = (half_t*)(W + oA + 128 * MB);
  half_t *FFH = (half_t*)(W + oA), *Z16 = (half_t*)(W + oA + 128 * MB), *GI = (half_t*)(W + oA);
  half_t *AO = (half_t*)(W + oAO), *Z3 = (half_t*)(W + oAO);
  float  *Z32 = (float*)(W + oZ32);
  half_t *HS0 = (half_t*)(W + oZ32), *HS1 = (half_t*)(W + oZ32 + 32 * MB);
  float  *FZ = (float*)(W + oFZ);
  half_t *HB = (half_t*)(W + oHB);
  unsigned int *CNT = (unsigned int*)(W + oCNT);

  auto cvt = [&](const float* s, half_t* d, int n) {
    cvt_f16_kernel<<<256, 256, 0, stream>>>(s, d, n);
  };
  // pack [Wq1;Wq2], [Wk1;Wk2], [Wv1;Wv2] as (512,512) f16; convert the rest
  cvt(fWq1, QW, 131072); cvt(fWq2, QW + 131072, 131072);
  cvt(fWk1, KW, 131072); cvt(fWk2, KW + 131072, 131072);
  cvt(fWv1, VW, 131072); cvt(fWv2, VW + 131072, 131072);
  cvt(fW0, W0h, 262144);
  cvt(fWf1, WF1, 2048 * 512); cvt(fWf2, WF2, 512 * 2048);
  cvt(fWih0, WIH0, 1536 * 512); cvt(fWhh0, WHH0, 1536 * 512);
  cvt(fWih1, WIH1, 1536 * 512); cvt(fWhh1, WHH1, 1536 * 512);

  gather_kernel<<<32768, 128, 0, stream>>>(cate, targ, Memb, Eemb, M16, E16);

  auto gemm = [&](const half_t* A, const half_t* Wm, const float* bias, float* Cf,
                  half_t* Ch, int M, int N, int K, int ldct, int mode) {
    gemm_wmma_kernel<<<dim3((M >> 6) * (N >> 6)), 256, 0, stream>>>(
        A, Wm, bias, Cf, Ch, M, N, K, ldct, mode);
  };
  gemm(E16, QW, nullptr, nullptr, Q16, 32768, 512, 512, 0, GEMM_OUT_F16);
  gemm(M16, KW, nullptr, nullptr, K16, 32768, 512, 512, 0, GEMM_OUT_F16);
  gemm(M16, VW, nullptr, nullptr, Vt16, 32768, 512, 512, 32768, GEMM_OUT_F16T);
  attn_kernel<<<1024, 128, 0, stream>>>(Q16, K16, Vt16, AO);
  gemm(AO, W0h, nullptr, Z32, Z16, 32768, 512, 512, 0, GEMM_OUT_F32 | GEMM_OUT_F16);
  gemm(Z16, WF1, bf1, nullptr, FFH, 32768, 2048, 512, 0, GEMM_RELU | GEMM_OUT_F16);
  gemm(FFH, WF2, bf2, FZ, nullptr, 32768, 512, 2048, 0, GEMM_OUT_F32);
  ln_kernel<<<4096, 256, 0, stream>>>(Z32, FZ, ln_g, ln_b, Z3);

  // GRU layer 0: gi = z3 @ Wih0^T + bih0, then cooperative recurrence
  gemm(Z3, WIH0, bih0, nullptr, GI, 32768, 1536, 512, 0, GEMM_OUT_F16);
  hipMemsetAsync(W + oHB, 0, 64 * 1024 + 4096, stream);
  gru_kernel<<<16, 128, 0, stream>>>(GI, WHH0, bhh0, HB, HS0, CNT);
  // GRU layer 1
  gemm(HS0, WIH1, bih1, nullptr, GI, 32768, 1536, 512, 0, GEMM_OUT_F16);
  hipMemsetAsync(W + oHB, 0, 64 * 1024 + 4096, stream);
  gru_kernel<<<16, 128, 0, stream>>>(GI, WHH1, bhh1, HB, HS1, CNT);

  head_kernel<<<4096, 256, 0, stream>>>(HS1, Wfin, bfin, (float*)d_out);
}
